// MultiHeadLatentAttention_33174327394330
// MI455X (gfx1250) — matrix operand order repaired and checked by reading, not compiled, not run on hardware
//
#include <hip/hip_runtime.h>
#include <hip/hip_bf16.h>

// ---------------------------------------------------------------------------
// Problem constants (from reference)
// ---------------------------------------------------------------------------
#define EMB      2048
#define HEADS    16
#define HEAD_DIM 128
#define LATENT   512
#define BATCH    2
#define SEQ      2048
#define MROWS    (BATCH * SEQ)      // 4096

typedef __bf16 bf16_t;
typedef __attribute__((ext_vector_type(16))) __bf16 v16bf;
typedef __attribute__((ext_vector_type(8)))  __bf16 v8bf;
typedef __attribute__((ext_vector_type(8)))  float  v8f;

// ---------------------------------------------------------------------------
// WMMA wrapper: D = A(16x32 bf16) * B(32x16 bf16) + C(16x16 f32)
// ---------------------------------------------------------------------------
__device__ __forceinline__ v8f wmma_bf16(v16bf a, v16bf b, v8f c) {
    return __builtin_amdgcn_wmma_f32_16x16x32_bf16(
        /*neg_a=*/false, a, /*neg_b=*/false, b,
        /*c_mod=*/(short)0, c, /*reuse_a=*/false, /*reuse_b=*/false);
}

// ---------------------------------------------------------------------------
// Fragment loaders, matching CDNA5 ISA VGPR layouts (CDNA5 ISA §7.12.2).
// A-matrix 16x32 (MxK), row-major source, ld in elements:
//   lane<16 : M=lane,    K = 0..7  (elems 0..7),  K = 16..23 (elems 8..15)
//   lane>=16: M=lane-16, K = 8..15 (elems 0..7),  K = 24..31 (elems 8..15)
// ---------------------------------------------------------------------------
__device__ __forceinline__ v16bf load_a_frag(const bf16_t* __restrict__ tile, int ld) {
    const int lane = threadIdx.x & 31;
    const int m  = lane & 15;
    const int kb = (lane >> 4) << 3;          // 0 or 8
    const bf16_t* p = tile + (size_t)m * ld + kb;
    union { v16bf v; v8bf h[2]; } u;
    u.h[0] = *(const v8bf*)(p);
    u.h[1] = *(const v8bf*)(p + 16);
    return u.v;
}

// B-matrix 32x16 (KxN) built from row-major W[N][K] (so B = W^T):
//   lane<16 : N=lane,    K = 0..7  (elems 0..7),  K = 16..23 (elems 8..15)
//   lane>=16: N=lane-16, K = 8..15 (elems 0..7),  K = 24..31 (elems 8..15)
__device__ __forceinline__ v16bf load_b_frag(const bf16_t* __restrict__ wtile, int ld) {
    const int lane = threadIdx.x & 31;
    const int n  = lane & 15;
    const int kb = (lane >> 4) << 3;          // 0 or 8
    const bf16_t* p = wtile + (size_t)n * ld + kb;
    union { v16bf v; v8bf h[2]; } u;
    u.h[0] = *(const v8bf*)(p);
    u.h[1] = *(const v8bf*)(p + 16);
    return u.v;
}

// ---------------------------------------------------------------------------
// Cross-lane reductions within 16-lane row groups via ds_swizzle
// (group-of-32 mode: offset = xor_mask<<10 | and_mask 0x1F)
// ---------------------------------------------------------------------------
__device__ __forceinline__ float swz_xor(float v, int imm_i) {
    // imm handled per-call below with constant offsets via template-free wrappers
    return v;
}
__device__ __forceinline__ float row16_max(float v) {
    v = fmaxf(v, __builtin_bit_cast(float, __builtin_amdgcn_ds_swizzle(__builtin_bit_cast(int, v), 0x041F)));
    v = fmaxf(v, __builtin_bit_cast(float, __builtin_amdgcn_ds_swizzle(__builtin_bit_cast(int, v), 0x081F)));
    v = fmaxf(v, __builtin_bit_cast(float, __builtin_amdgcn_ds_swizzle(__builtin_bit_cast(int, v), 0x101F)));
    v = fmaxf(v, __builtin_bit_cast(float, __builtin_amdgcn_ds_swizzle(__builtin_bit_cast(int, v), 0x201F)));
    return v;
}
__device__ __forceinline__ float row16_sum(float v) {
    v += __builtin_bit_cast(float, __builtin_amdgcn_ds_swizzle(__builtin_bit_cast(int, v), 0x041F));
    v += __builtin_bit_cast(float, __builtin_amdgcn_ds_swizzle(__builtin_bit_cast(int, v), 0x081F));
    v += __builtin_bit_cast(float, __builtin_amdgcn_ds_swizzle(__builtin_bit_cast(int, v), 0x101F));
    v += __builtin_bit_cast(float, __builtin_amdgcn_ds_swizzle(__builtin_bit_cast(int, v), 0x201F));
    return v;
}

// ---------------------------------------------------------------------------
// f32 -> bf16 conversion (vectorized x4, n must be a multiple of 4)
// ---------------------------------------------------------------------------
__global__ __launch_bounds__(256) void cvt_f32_bf16(const float* __restrict__ in,
                                                    bf16_t* __restrict__ out,
                                                    size_t n4) {
    size_t i = (size_t)blockIdx.x * blockDim.x + threadIdx.x;
    if (i >= n4) return;
    float4 f = ((const float4*)in)[i];
    bf16_t o4[4] = { (bf16_t)f.x, (bf16_t)f.y, (bf16_t)f.z, (bf16_t)f.w };
    *(unsigned long long*)(out + i * 4) = *(unsigned long long*)o4;
}

// ---------------------------------------------------------------------------
// GEMM: C[M,N] = A[M,K] @ W[N,K]^T   (torch Linear convention)
// Block: 256 threads = 8 waves, arranged 2(M) x 4(N); block tile 64x256.
// Wave tile: 32x64 (2 M-tiles x 4 N-tiles of 16x16).
// mode 0: store bf16 row-major into Cbf
// mode 1: store f32 + bias into Cf  (final projection)
// mode 2: store bf16 transposed-per-head into Vt[b][h][d][s]
// ---------------------------------------------------------------------------
__global__ __launch_bounds__(256) void gemm_bf16_wmma(
    const bf16_t* __restrict__ A, int lda,
    const bf16_t* __restrict__ W, int ldw,
    bf16_t* __restrict__ Cbf, float* __restrict__ Cf,
    const float* __restrict__ bias,
    bf16_t* __restrict__ Vt, int Slen,
    int N, int K, int mode)
{
    const int wave = threadIdx.x >> 5;        // 0..7
    const int wm   = wave >> 2;               // 0..1
    const int wn   = wave & 3;                // 0..3
    const int m0   = blockIdx.x * 64  + wm * 32;
    const int n0   = blockIdx.y * 256 + wn * 64;

    v8f acc[2][4];
#pragma unroll
    for (int i = 0; i < 2; ++i)
#pragma unroll
        for (int t = 0; t < 4; ++t) acc[i][t] = (v8f){};

    const bf16_t* a0p = A + (size_t)m0 * lda;
    const bf16_t* a1p = A + (size_t)(m0 + 16) * lda;

    for (int k0 = 0; k0 < K; k0 += 32) {
        if (k0 + 32 < K) {
            __builtin_prefetch(a0p + k0 + 32, 0, 1);
            __builtin_prefetch(a1p + k0 + 32, 0, 1);
        }
        v16bf af0 = load_a_frag(a0p + k0, lda);
        v16bf af1 = load_a_frag(a1p + k0, lda);
#pragma unroll
        for (int t = 0; t < 4; ++t) {
            v16bf bf = load_b_frag(W + (size_t)(n0 + t * 16) * ldw + k0, ldw);
            acc[0][t] = wmma_bf16(af0, bf, acc[0][t]);
            acc[1][t] = wmma_bf16(af1, bf, acc[1][t]);
        }
    }

    // Epilogue. C-layout: element r of v8f, lane l -> row = r + 8*(l>>4), col = l&15
    const int lane  = threadIdx.x & 31;
    const int col   = lane & 15;
    const int rbase = (lane >> 4) << 3;

#pragma unroll
    for (int mi = 0; mi < 2; ++mi) {
#pragma unroll
        for (int t = 0; t < 4; ++t) {
#pragma unroll
            for (int r = 0; r < 8; ++r) {
                const int m = m0 + mi * 16 + r + rbase;
                const int n = n0 + t * 16 + col;
                const float val = acc[mi][t][r];
                if (mode == 0) {
                    Cbf[(size_t)m * N + n] = (bf16_t)val;
                } else if (mode == 1) {
                    Cf[(size_t)m * N + n] = val + bias[n];
                } else {
                    const int b = m / Slen, s = m % Slen;
                    const int h = n >> 7,  d = n & 127;
                    Vt[(((size_t)b * HEADS + h) * HEAD_DIM + d) * Slen + s] = (bf16_t)val;
                }
            }
        }
    }
}

// ---------------------------------------------------------------------------
// Causal attention, flash-style. One wave per 16-row query tile.
// q,k: bf16 [B,S,EMB] (head slice has row stride EMB, contiguous d)
// vT : bf16 [B,H,HEAD_DIM,S]  -> ctx = P @ (vT)^T, same A@W^T pattern
// ctx: bf16 [B,S,EMB]
// ---------------------------------------------------------------------------
__global__ __launch_bounds__(128) void mla_attention(
    const bf16_t* __restrict__ q,
    const bf16_t* __restrict__ k,
    const bf16_t* __restrict__ vT,
    bf16_t* __restrict__ ctx)
{
    __shared__ bf16_t pbuf[4][16][32];        // per-wave P tile (16 q x 32 keys)

    const int wave = threadIdx.x >> 5;
    const int lane = threadIdx.x & 31;
    const int col  = lane & 15;
    const int rbase = (lane >> 4) << 3;

    const int idx = blockIdx.x * 4 + wave;    // 0 .. B*H*(S/16)-1
    const int qi  = idx & (SEQ / 16 - 1);     // query tile (0..127)
    const int h   = (idx >> 7) & (HEADS - 1);
    const int b   = idx >> 11;

    const bf16_t* qh = q  + (size_t)b * SEQ * EMB + (size_t)h * HEAD_DIM;
    const bf16_t* kh = k  + (size_t)b * SEQ * EMB + (size_t)h * HEAD_DIM;
    const bf16_t* vh = vT + (((size_t)b * HEADS + h) * HEAD_DIM) * SEQ;

    // Q fragments: 4 chunks of K-dim 32 covering D=128
    v16bf qa[4];
#pragma unroll
    for (int c = 0; c < 4; ++c)
        qa[c] = load_a_frag(qh + (size_t)(qi * 16) * EMB + c * 32, EMB);

    v8f acc[8];
#pragma unroll
    for (int t = 0; t < 8; ++t) acc[t] = (v8f){};
    float mrun[8], lrun[8];
#pragma unroll
    for (int r = 0; r < 8; ++r) { mrun[r] = -1e30f; lrun[r] = 0.0f; }

    const float scale = 0.0883883476483f;     // 1/sqrt(128)

    for (int jt = 0; jt <= qi; jt += 2) {     // process 32 keys per step
        v8f sc[2];
#pragma unroll
        for (int u = 0; u < 2; ++u) {
            const int j = jt + u;
            if (j <= qi) {
                v8f s = (v8f){};
#pragma unroll
                for (int c = 0; c < 4; ++c) {
                    v16bf kf = load_b_frag(kh + (size_t)(j * 16) * EMB + c * 32, EMB);
                    s = wmma_bf16(qa[c], kf, s);
                }
#pragma unroll
                for (int r = 0; r < 8; ++r) {
                    const int sq = qi * 16 + r + rbase;
                    const int sk = j * 16 + col;
                    const float v = s[r] * scale;
                    sc[u][r] = (sk <= sq) ? v : -1e30f;
                }
            } else {
#pragma unroll
                for (int r = 0; r < 8; ++r) sc[u][r] = -1e30f;
            }
        }

        // Online softmax over this 32-key block
        float f[8];
#pragma unroll
        for (int r = 0; r < 8; ++r) {
            float mx = row16_max(fmaxf(sc[0][r], sc[1][r]));
            const float mnew = fmaxf(mrun[r], mx);
            const float p0 = __expf(sc[0][r] - mnew);
            const float p1 = __expf(sc[1][r] - mnew);
            const float rs = row16_sum(p0 + p1);
            f[r] = __expf(mrun[r] - mnew);
            lrun[r] = lrun[r] * f[r] + rs;
            mrun[r] = mnew;
            const int row = r + rbase;
            pbuf[wave][row][col]      = (bf16_t)p0;
            pbuf[wave][row][col + 16] = (bf16_t)p1;
        }
        asm volatile("s_wait_dscnt 0" ::: "memory");

        // Rescale running accumulator, then ctx += P @ V over 8 d-tiles
#pragma unroll
        for (int t = 0; t < 8; ++t)
#pragma unroll
            for (int r = 0; r < 8; ++r) acc[t][r] *= f[r];

        v16bf pa = load_a_frag(&pbuf[wave][0][0], 32);
#pragma unroll
        for (int t = 0; t < 8; ++t) {
            v16bf vf = load_b_frag(vh + (size_t)(t * 16) * SEQ + jt * 16, SEQ);
            acc[t] = wmma_bf16(pa, vf, acc[t]);
        }
    }

    // Normalize and store ctx tile
#pragma unroll
    for (int r = 0; r < 8; ++r) {
        const float inv = 1.0f / lrun[r];
        const int s = qi * 16 + r + rbase;
#pragma unroll
        for (int t = 0; t < 8; ++t) {
            const int d = t * 16 + col;
            ctx[((size_t)b * SEQ + s) * EMB + h * HEAD_DIM + d] = (bf16_t)(acc[t][r] * inv);
        }
    }
}

// ---------------------------------------------------------------------------
// Host-side orchestration
// ---------------------------------------------------------------------------
extern "C" void kernel_launch(void* const* d_in, const int* in_sizes, int n_in,
                              void* d_out, int out_size, void* d_ws, size_t ws_size,
                              hipStream_t stream) {
    (void)in_sizes; (void)n_in; (void)out_size; (void)ws_size;

    const float* x      = (const float*)d_in[0];  // [B,S,EMB]
    const float* w_q    = (const float*)d_in[1];  // [EMB,EMB]
    const float* w_down = (const float*)d_in[2];  // [LATENT,EMB]
    const float* w_up_k = (const float*)d_in[3];  // [EMB,LATENT]
    const float* w_up_v = (const float*)d_in[4];  // [EMB,LATENT]
    const float* w_out  = (const float*)d_in[5];  // [EMB,EMB]
    const float* b_out  = (const float*)d_in[6];  // [EMB]
    float* out = (float*)d_out;

    size_t off = 0;
    auto alloc = [&](size_t bytes) -> void* {
        void* p = (char*)d_ws + off;
        off += (bytes + 255) & ~(size_t)255;
        return p;
    };
    bf16_t* xb   = (bf16_t*)alloc((size_t)MROWS * EMB * 2);
    bf16_t* wqb  = (bf16_t*)alloc((size_t)EMB * EMB * 2);
    bf16_t* wdb  = (bf16_t*)alloc((size_t)LATENT * EMB * 2);
    bf16_t* wukb = (bf16_t*)alloc((size_t)EMB * LATENT * 2);
    bf16_t* wuvb = (bf16_t*)alloc((size_t)EMB * LATENT * 2);
    bf16_t* wob  = (bf16_t*)alloc((size_t)EMB * EMB * 2);
    bf16_t* qb   = (bf16_t*)alloc((size_t)MROWS * EMB * 2);
    bf16_t* latb = (bf16_t*)alloc((size_t)MROWS * LATENT * 2);
    bf16_t* kb   = (bf16_t*)alloc((size_t)MROWS * EMB * 2);
    bf16_t* vTb  = (bf16_t*)alloc((size_t)BATCH * HEADS * HEAD_DIM * SEQ * 2);
    bf16_t* ctxb = (bf16_t*)alloc((size_t)MROWS * EMB * 2);

    auto cvt = [&](const float* src, bf16_t* dst, size_t n) {
        size_t n4 = n / 4;
        cvt_f32_bf16<<<(unsigned)((n4 + 255) / 256), 256, 0, stream>>>(src, dst, n4);
    };
    cvt(x,      xb,   (size_t)MROWS * EMB);
    cvt(w_q,    wqb,  (size_t)EMB * EMB);
    cvt(w_down, wdb,  (size_t)LATENT * EMB);
    cvt(w_up_k, wukb, (size_t)EMB * LATENT);
    cvt(w_up_v, wuvb, (size_t)EMB * LATENT);
    cvt(w_out,  wob,  (size_t)EMB * EMB);

    // q = x @ w_q^T            [4096,2048] = [4096,2048] x [2048,2048]^T
    gemm_bf16_wmma<<<dim3(MROWS / 64, EMB / 256), 256, 0, stream>>>(
        xb, EMB, wqb, EMB, qb, nullptr, nullptr, nullptr, SEQ, EMB, EMB, 0);
    // latent = x @ w_down^T    [4096,512]
    gemm_bf16_wmma<<<dim3(MROWS / 64, LATENT / 256), 256, 0, stream>>>(
        xb, EMB, wdb, EMB, latb, nullptr, nullptr, nullptr, SEQ, LATENT, EMB, 0);
    // k = latent @ w_up_k^T    [4096,2048]
    gemm_bf16_wmma<<<dim3(MROWS / 64, EMB / 256), 256, 0, stream>>>(
        latb, LATENT, wukb, LATENT, kb, nullptr, nullptr, nullptr, SEQ, EMB, LATENT, 0);
    // v = latent @ w_up_v^T, stored transposed per head -> vT[b][h][d][s]
    gemm_bf16_wmma<<<dim3(MROWS / 64, EMB / 256), 256, 0, stream>>>(
        latb, LATENT, wuvb, LATENT, nullptr, nullptr, nullptr, vTb, SEQ, EMB, LATENT, 2);

    // attention: B*H*(S/16) = 4096 waves, 4 waves per block
    mla_attention<<<(BATCH * HEADS * (SEQ / 16)) / 4, 128, 0, stream>>>(qb, kb, vTb, ctxb);

    // out = ctx @ w_out^T + b_out  (f32 output)
    gemm_bf16_wmma<<<dim3(MROWS / 64, EMB / 256), 256, 0, stream>>>(
        ctxb, EMB, wob, EMB, nullptr, out, b_out, nullptr, SEQ, EMB, EMB, 1);
}
